// dfe_module_31834297598157
// MI455X (gfx1250) — matrix operand order repaired and checked by reading, not compiled, not run on hardware
//
#include <hip/hip_runtime.h>

#define CIN 512
#define CC  256
#define SS  16384
#define NN  4
#define NSPLIT 16

typedef unsigned short bf16_t;
typedef __attribute__((ext_vector_type(16))) __bf16       v16bf;
typedef __attribute__((ext_vector_type(8)))  float        v8f;
typedef __attribute__((ext_vector_type(4)))  unsigned int v4u;
typedef __attribute__((ext_vector_type(8)))  unsigned int v8u;

// LDS row padding (in bf16 elements) to spread banks
#define PAD1 (CIN + 8)   // conv1 feat tile rows
#define PAD2 (CC  + 8)   // q / out tile rows

static __device__ __forceinline__ bf16_t f2bf(float f) {
  unsigned int u = __builtin_bit_cast(unsigned int, f);
  u += 0x7FFFu + ((u >> 16) & 1u);     // round-to-nearest-even
  return (bf16_t)(u >> 16);
}

// A fragment (16x32 bf16, M=lane%16): two contiguous 16B chunks at K = 16h / 16h+8
static __device__ __forceinline__ v16bf load_a_frag(const bf16_t* p) {
  v4u lo = *(const v4u*)(p);
  v4u hi = *(const v4u*)(p + 8);
  v8u v = { lo[0], lo[1], lo[2], lo[3], hi[0], hi[1], hi[2], hi[3] };
  return __builtin_bit_cast(v16bf, v);
}

// B fragment (32x16 bf16, N=lane%16): one contiguous 32B chunk at K = 16h
static __device__ __forceinline__ v16bf load_b_frag(const bf16_t* p) {
  v8u v = *(const v8u*)(p);
  return __builtin_bit_cast(v16bf, v);
}

static __device__ __forceinline__ v8f wmma_bf16(v16bf a, v16bf b, v8f c) {
  return __builtin_amdgcn_wmma_f32_16x16x32_bf16(false, a, false, b, (short)0, c,
                                                 false, false);
}

// ---------------------------------------------------------------- prep kernels
__global__ void __launch_bounds__(256)
prep_weights(const float* __restrict__ w1, const float* __restrict__ w2,
             const float* __restrict__ gamma, const float* __restrict__ beta,
             const float* __restrict__ mean, const float* __restrict__ var,
             bf16_t* __restrict__ w1b, bf16_t* __restrict__ w2b,
             float* __restrict__ scale, float* __restrict__ bias) {
  int i = blockIdx.x * 256 + threadIdx.x;
  if (i < CIN * CC) w1b[i] = f2bf(w1[i]);
  if (i < CC * CC)  w2b[i] = f2bf(w2[i]);
  if (i < CC) {
    float sc = gamma[i] * rsqrtf(var[i] + 1e-5f);
    scale[i] = sc;
    bias[i]  = beta[i] - mean[i] * sc;
  }
}

__global__ void __launch_bounds__(256)
convert_q(const float* __restrict__ x, bf16_t* __restrict__ qb) {
  size_t i = (size_t)blockIdx.x * 256 + threadIdx.x;   // 4 elems per thread
  const float4* in4 = (const float4*)x;
  float4 v = in4[i];
  unsigned int p0 = (unsigned int)f2bf(v.x) | ((unsigned int)f2bf(v.y) << 16);
  unsigned int p1 = (unsigned int)f2bf(v.z) | ((unsigned int)f2bf(v.w) << 16);
  uint2 p; p.x = p0; p.y = p1;
  ((uint2*)qb)[i] = p;
}

// ------------------------------------------- stage 1: conv1 + BN + ReLU -> k_bf
// grid: N * (SS/16) blocks of 256 threads; block computes all 256 o for one s-tile
__global__ void __launch_bounds__(256)
conv1_bn_relu(const float* __restrict__ feat, const bf16_t* __restrict__ w1b,
              const float* __restrict__ scale, const float* __restrict__ bias,
              bf16_t* __restrict__ kbf) {
  __shared__ bf16_t ldsB[16 * PAD1];          // feat tile transposed: [s][i]
  int blk = blockIdx.x;
  int n   = blk >> 10;
  int s0  = (blk & 1023) << 4;
  int tid = threadIdx.x;

  const float* fbase = feat + (size_t)n * CIN * SS + s0;
  // phase 0: load + convert feat[n, :, s0:s0+16] into LDS transposed
  for (int base = 0; base < CIN; base += 16) {
    int i = base + (tid >> 4);
    int s = tid & 15;
    ldsB[s * PAD1 + i] = f2bf(fbase[(size_t)i * SS + s]);
  }
  __syncthreads();

  int wave = tid >> 5, lane = tid & 31;
  int l0 = lane & 15, h = lane >> 4;
  const bf16_t* brow = ldsB + l0 * PAD1 + 16 * h;

  for (int t = 0; t < 2; ++t) {
    int o0 = wave * 32 + t * 16;
    v8f acc = {};
    const bf16_t* arow = w1b + (size_t)(o0 + l0) * CIN + 16 * h;
#pragma unroll 4
    for (int k = 0; k < CIN; k += 32) {
      v16bf a = load_a_frag(arow + k);
      v16bf b = load_b_frag(brow + k);
      acc = wmma_bf16(a, b, acc);
    }
#pragma unroll
    for (int r = 0; r < 8; ++r) {
      int o = o0 + r + 8 * h;
      float v = acc[r] * scale[o] + bias[o];
      v = v > 0.0f ? v : 0.0f;
      kbf[((size_t)n * CC + o) * SS + s0 + l0] = f2bf(v);
    }
  }
}

// ------------------------------------------------ stage 2: energy (split-K GEMM)
// one wave per (n, d-tile, c-tile, split); each split writes a private partial
// slice (no atomics -> deterministic, plain stores)
__global__ void __launch_bounds__(256)
energy_gemm(const bf16_t* __restrict__ qbf, const bf16_t* __restrict__ kbf,
            float* __restrict__ epart) {
  int gw = blockIdx.x * 8 + (threadIdx.x >> 5);
  int lane = threadIdx.x & 31;
  int l0 = lane & 15, h = lane >> 4;
  int split = gw & 15, ct = (gw >> 4) & 15, dt = (gw >> 8) & 15, n = gw >> 12;

  const bf16_t* arow = qbf + ((size_t)n * CC + dt * 16 + l0) * SS + 16 * h;
  const bf16_t* brow = kbf + ((size_t)n * CC + ct * 16 + l0) * SS + 16 * h;
  int sbeg = split * (SS / NSPLIT);

  v8f acc = {};
#pragma unroll 4
  for (int k = sbeg; k < sbeg + SS / NSPLIT; k += 32) {
    v16bf a = load_a_frag(arow + k);
    v16bf b = load_b_frag(brow + k);
    acc = wmma_bf16(a, b, acc);
  }
  float* ebase = epart + (size_t)split * (NN * CC * CC)
               + (size_t)n * CC * CC + ct * 16 + l0;
#pragma unroll
  for (int r = 0; r < 8; ++r) {
    int d = dt * 16 + r + 8 * h;
    ebase[(size_t)d * CC] = acc[r];
  }
}

__global__ void __launch_bounds__(256)
reduce_energy(const float* __restrict__ epart, float* __restrict__ energy) {
  size_t i = (size_t)blockIdx.x * 256 + threadIdx.x;
  float s = 0.0f;
#pragma unroll
  for (int j = 0; j < NSPLIT; ++j)
    s += epart[(size_t)j * (NN * CC * CC) + i];
  energy[i] = s;
}

// ----------------------------- stage 2b: softmax(max - e) and transposed store
// softmax(max - e) == exp(min(e) - e) / sum; block = one (n,d) row of 256
__global__ void __launch_bounds__(256)
softmax_t(const float* __restrict__ energy, bf16_t* __restrict__ attT) {
  __shared__ float red[256];
  int n = blockIdx.x >> 8, d = blockIdx.x & 255;
  int c = threadIdx.x;
  float e = energy[((size_t)n * CC + d) * CC + c];

  red[c] = e; __syncthreads();
  for (int off = 128; off > 0; off >>= 1) {
    if (c < off) red[c] = fminf(red[c], red[c + off]);
    __syncthreads();
  }
  float emin = red[0];
  __syncthreads();
  float p = __expf(emin - e);
  red[c] = p; __syncthreads();
  for (int off = 128; off > 0; off >>= 1) {
    if (c < off) red[c] += red[c + off];
    __syncthreads();
  }
  float inv = 1.0f / red[0];
  attT[((size_t)n * CC + c) * CC + d] = f2bf(p * inv);   // store A^T for stage 3
}

// ------------------- stage 3+4 fused: out = att^T @ q ; final = w2 @ out + b2
// grid: N * (SS/16); q-tile -> LDS (transposed), GEMM1 -> LDS, GEMM2 -> d_out
__global__ void __launch_bounds__(256)
attn_conv2(const bf16_t* __restrict__ attT, const bf16_t* __restrict__ qbf,
           const bf16_t* __restrict__ w2b, const float* __restrict__ b2,
           float* __restrict__ out) {
  __shared__ bf16_t ldsQ[16 * PAD2];   // q tile transposed: [s][d]
  __shared__ bf16_t ldsO[16 * PAD2];   // out tile transposed: [s][c]
  int blk = blockIdx.x;
  int n   = blk >> 10;
  int s0  = (blk & 1023) << 4;
  int tid = threadIdx.x;

  const bf16_t* qb = qbf + (size_t)n * CC * SS + s0;
  for (int base = 0; base < CC; base += 16) {
    int d = base + (tid >> 4);
    int s = tid & 15;
    ldsQ[s * PAD2 + d] = qb[(size_t)d * SS + s];
  }
  __syncthreads();

  int wave = tid >> 5, lane = tid & 31;
  int l0 = lane & 15, h = lane >> 4;

  // phase 1: out[c, s-tile] = sum_d attT[c,d] * q[d,s]
  const bf16_t* browQ = ldsQ + l0 * PAD2 + 16 * h;
  for (int t = 0; t < 2; ++t) {
    int c0 = wave * 32 + t * 16;
    v8f acc = {};
    const bf16_t* arow = attT + ((size_t)n * CC + c0 + l0) * CC + 16 * h;
#pragma unroll
    for (int k = 0; k < CC; k += 32) {
      v16bf a = load_a_frag(arow + k);
      v16bf b = load_b_frag(browQ + k);
      acc = wmma_bf16(a, b, acc);
    }
#pragma unroll
    for (int r = 0; r < 8; ++r) {
      int cc = c0 + r + 8 * h;
      ldsO[l0 * PAD2 + cc] = f2bf(acc[r]);
    }
  }
  __syncthreads();

  // phase 2: final[o, s-tile] = sum_c w2[o,c] * out[c,s] + b2[o]
  const bf16_t* browO = ldsO + l0 * PAD2 + 16 * h;
  for (int t = 0; t < 2; ++t) {
    int o0 = wave * 32 + t * 16;
    v8f acc = {};
    const bf16_t* arow = w2b + (size_t)(o0 + l0) * CC + 16 * h;
#pragma unroll
    for (int k = 0; k < CC; k += 32) {
      v16bf a = load_a_frag(arow + k);
      v16bf b = load_b_frag(browO + k);
      acc = wmma_bf16(a, b, acc);
    }
#pragma unroll
    for (int r = 0; r < 8; ++r) {
      int o = o0 + r + 8 * h;
      out[((size_t)n * CC + o) * SS + s0 + l0] = acc[r] + b2[o];
    }
  }
}

// ---------------------------------------------------------------------- launch
extern "C" void kernel_launch(void* const* d_in, const int* in_sizes, int n_in,
                              void* d_out, int out_size, void* d_ws, size_t ws_size,
                              hipStream_t stream) {
  (void)in_sizes; (void)n_in; (void)out_size; (void)ws_size;
  const float* feat   = (const float*)d_in[0];
  const float* coarse = (const float*)d_in[1];
  const float* w1     = (const float*)d_in[2];
  const float* gamma  = (const float*)d_in[3];
  const float* beta   = (const float*)d_in[4];
  const float* mean   = (const float*)d_in[5];
  const float* var    = (const float*)d_in[6];
  const float* w2     = (const float*)d_in[7];
  const float* b2     = (const float*)d_in[8];
  float* out = (float*)d_out;

  size_t off = 0;
  char* base = (char*)d_ws;
  auto alloc = [&](size_t bytes) -> void* {
    void* p = base + off;
    off += (bytes + 255) & ~(size_t)255;
    return p;
  };
  bf16_t* w1b   = (bf16_t*)alloc((size_t)CIN * CC * 2);
  bf16_t* w2b   = (bf16_t*)alloc((size_t)CC * CC * 2);
  float*  scale = (float*) alloc((size_t)CC * 4);
  float*  bias  = (float*) alloc((size_t)CC * 4);
  bf16_t* qbf   = (bf16_t*)alloc((size_t)NN * CC * SS * 2);
  bf16_t* kbf   = (bf16_t*)alloc((size_t)NN * CC * SS * 2);
  float*  epart = (float*) alloc((size_t)NSPLIT * NN * CC * CC * 4);
  float*  energ = (float*) alloc((size_t)NN * CC * CC * 4);
  bf16_t* attT  = (bf16_t*)alloc((size_t)NN * CC * CC * 2);

  prep_weights<<<(CIN * CC) / 256, 256, 0, stream>>>(w1, w2, gamma, beta, mean,
                                                     var, w1b, w2b, scale, bias);
  convert_q<<<(NN * CC * (SS / 4)) / 256, 256, 0, stream>>>(coarse, qbf);
  conv1_bn_relu<<<NN * (SS / 16), 256, 0, stream>>>(feat, w1b, scale, bias, kbf);
  energy_gemm<<<(NN * 16 * 16 * NSPLIT) / 8, 256, 0, stream>>>(qbf, kbf, epart);
  reduce_energy<<<(NN * CC * CC) / 256, 256, 0, stream>>>(epart, energ);
  softmax_t<<<NN * CC, 256, 0, stream>>>(energ, attT);
  attn_conv2<<<NN * (SS / 16), 256, 0, stream>>>(attT, qbf, w2b, b2, out);
}
